// DNCLSTM_11630771438087
// MI455X (gfx1250) — compile-verified, matrix operand order checked by reading, not run on hardware
//
#include <hip/hip_runtime.h>
#include <hip/hip_bf16.h>

typedef __attribute__((ext_vector_type(16))) __bf16 v16bf;
typedef __attribute__((ext_vector_type(8)))  __bf16 v8bf;
typedef __attribute__((ext_vector_type(8)))  float  v8f;

#define Tn   128
#define Bn   64
#define Dn   256
#define Hn   512
#define Rn   4
#define Nn   128
#define Wn   64
#define Kn   1024                 // GEMM K = (D + R*W) + H = 512 + 512
#define Gn   2048                 // 4*H gate columns
#define B16  16                   // batch rows per workgroup (== WMMA M)
#define NWG  (Bn / B16)           // 4 workgroups
#define NTHREADS 512
#define NWAVES  (NTHREADS / 32)   // 16 waves
#define NTILES  (Gn / 16)         // 128 N-tiles
#define TPW     (NTILES / NWAVES) // 8 tiles per wave
#define KBLK    (Kn / 32)         // 32 k-blocks of 32
#define OUTC    (Hn + Wn * Rn)    // 768

// workspace layout (bytes)
#define WS_WFRAG_OFF  0u                      // bf16 [Gn*Kn]            = 4 MB
#define WS_MEM_OFF    4194304u                // f32  [Bn*Nn*Wn]         = 2 MB
#define WS_LINK_OFF   6291456u                // f32  [Bn*Nn*Nn]         = 4 MB

__device__ __forceinline__ float sigm(float v) { return 1.f / (1.f + expf(-v)); }
__device__ __forceinline__ float oneplus(float v) {
  // 1 + softplus(v), numerically stable
  return 1.f + fmaxf(v, 0.f) + log1pf(expf(-fabsf(v)));
}
__device__ __forceinline__ float dot512(const float* __restrict__ w, const float* h) {
  float s = 0.f;
#pragma unroll 8
  for (int j = 0; j < Hn; ++j) s = fmaf(w[j], h[j], s);
  return s;
}

// ---------------------------------------------------------------------------
// Prep: swizzle [W_ih | W_hh] (fp32, row-major [2048 x 512] each) into bf16
// WMMA B-fragments. Fragment (tile, kb) holds the 32(K) x 16(N) block:
// element index = lane*16 + i, with
//   n = tile*16 + (lane & 15)
//   k = kb*32 + (lane>>4)*8 + (i < 8 ? i : i + 8)
// matching the CDNA5 16-bit operand VGPR layout.
// ---------------------------------------------------------------------------
__global__ void dnc_prep(const float* __restrict__ W_ih,
                         const float* __restrict__ W_hh,
                         __bf16* __restrict__ Wfrag) {
  int idx = blockIdx.x * blockDim.x + threadIdx.x;
  if (idx >= Gn * Kn) return;
  int frag = idx >> 9;            // 512 elements per fragment
  int e    = idx & 511;
  int lane = e >> 4;
  int i    = e & 15;
  int tile = frag >> 5;           // / KBLK
  int kb   = frag & 31;
  int n = tile * 16 + (lane & 15);
  int k = kb * 32 + ((lane >> 4) << 3) + (i < 8 ? i : i + 8);
  float v = (k < 512) ? W_ih[n * 512 + k] : W_hh[n * 512 + (k - 512)];
  Wfrag[idx] = (__bf16)v;
}

// ---------------------------------------------------------------------------
// Main persistent DNC kernel: 1 workgroup == 16 batch rows, full T-step
// recurrence on-chip. 512 threads = 16 waves; GEMM M=16, N=2048, K=1024
// via v_wmma_f32_16x16x32_bf16 (8 tiles/wave, 32 k-blocks).
// ---------------------------------------------------------------------------
__global__ void __launch_bounds__(NTHREADS, 1)
dnc_forward(const float* __restrict__ x,   const float* __restrict__ M0,
            const float* __restrict__ b_ih, const float* __restrict__ b_hh,
            const float* __restrict__ Wrk, const float* __restrict__ brk,
            const float* __restrict__ Wrs, const float* __restrict__ brs,
            const float* __restrict__ Wwk, const float* __restrict__ bwk,
            const float* __restrict__ Wws, const float* __restrict__ bws,
            const float* __restrict__ We,  const float* __restrict__ be,
            const float* __restrict__ Wv,  const float* __restrict__ bv,
            const float* __restrict__ Wfg, const float* __restrict__ bfg,
            const float* __restrict__ Wag, const float* __restrict__ bag,
            const float* __restrict__ Wwg, const float* __restrict__ bwg,
            const float* __restrict__ Wrm, const float* __restrict__ brm,
            const __bf16* __restrict__ Wfrag,
            float* __restrict__ memG, float* __restrict__ linkG,
            float* __restrict__ out) {
  // ---- LDS state (~290 KB of the 320 KB WGP LDS) ----
  __shared__ __align__(16) __bf16 sA[B16][Kn];        // 32 KB  A = [x_t | r | h] bf16
  __shared__ float sG[B16 * Gn];                      // 128 KB gates + scratch pool
  __shared__ float sH[B16][Hn];                       // 32 KB
  __shared__ float sS[B16][Hn];                       // 32 KB
  __shared__ float sRW[B16][Nn][Rn];                  // 32 KB  read weights (carry)
  __shared__ float sWW[B16][Nn];                      // 8 KB   write weights (carry)
  __shared__ float sU[B16][Nn];                       // 8 KB   usage
  __shared__ float sP[B16][Nn];                       // 8 KB   precedence
  __shared__ float sBias[Gn];                         // 8 KB   b_ih + b_hh
  __shared__ float sFG[B16][Rn], sRS[B16][Rn], sRM[B16][3][Rn];
  __shared__ float sWSs[B16], sAG[B16], sWGt[B16], sKN[B16][5], sWSum[B16];

  const int tid  = threadIdx.x;
  const int lane = tid & 31;
  const int wave = tid >> 5;
  const int b0   = blockIdx.x * B16;

  float* memB  = memG  + (size_t)b0 * Nn * Wn;        // [B16][Nn][Wn]
  float* linkB = linkG + (size_t)b0 * Nn * Nn;        // [B16][Nn][Nn]

  // scratch overlay inside sG (units: floats; pool = 32768 floats)
  float* scr = sG;
  const int RK_OFF  = 0;       // [16][64][4]   4096
  const int WK_OFF  = 4096;    // [16][64]      1024
  const int ER_OFF  = 5120;    // [16][64]      1024
  const int WV_OFF  = 6144;    // [16][64]      1024
  const int SC_OFF  = 7168;    // [16][128][5] 10240 (r=0..3 read, 4 write; softmax in place)
  const int AL_OFF  = 17408;   // [16][128]     2048 allocation (dead before RV)
  const int SU_OFF  = 19456;   // [16][128]     2048 sorted usage (dead before RV)
  const int RWN_OFF = 21504;   // [16][128][4]  8192 new read weights
  const int RV_OFF  = 17408;   // [16][256]     4096 read vectors (overlays AL+SU)

  // ---- init carries / memory / bias ----
  for (int i = tid; i < Gn; i += NTHREADS) sBias[i] = b_ih[i] + b_hh[i];
  for (int i = tid; i < B16 * Hn; i += NTHREADS) { (&sH[0][0])[i] = 0.f; (&sS[0][0])[i] = 0.f; }
  for (int i = tid; i < B16 * Nn * Rn; i += NTHREADS) (&sRW[0][0][0])[i] = 0.f;
  for (int i = tid; i < B16 * Nn; i += NTHREADS) {
    (&sWW[0][0])[i] = 0.f; (&sU[0][0])[i] = 1e-6f; (&sP[0][0])[i] = 0.f;
  }
  for (int i = tid; i < B16 * (Kn - Dn); i += NTHREADS) {  // r and h columns start at 0
    int m = i / (Kn - Dn), c = i % (Kn - Dn);
    sA[m][Dn + c] = (__bf16)0.f;
  }
  for (int i = tid; i < B16 * Nn * Wn; i += NTHREADS) memB[i] = M0[i % (Nn * Wn)];
  for (int i = tid; i < B16 * Nn * Nn; i += NTHREADS) linkB[i] = 0.f;
  __syncthreads();

  for (int t = 0; t < Tn; ++t) {
    // (a) x_t -> sA[:, 0:256] (bf16)
    for (int i = tid; i < B16 * Dn; i += NTHREADS) {
      int m = i >> 8, k = i & 255;
      sA[m][k] = (__bf16)x[((size_t)t * Bn + (b0 + m)) * Dn + k];
    }
    __syncthreads();

    // (b) gate GEMM: [16,1024] x [1024,2048] -> sG, bf16 WMMA, fp32 acc.
    // Per k-block: batch all 8 B-fragment loads (one clause), then 8 WMMAs,
    // so in-order load returns overlap matrix execution.
    {
      v8f acc[TPW];
      const v8f vz = {0.f, 0.f, 0.f, 0.f, 0.f, 0.f, 0.f, 0.f};
#pragma unroll
      for (int q = 0; q < TPW; ++q) acc[q] = vz;

      const int mrow  = lane & 15;
      const int khalf = (lane >> 4) << 3;  // 0 or 8
      const __bf16* Arow = &sA[mrow][0];
      const __bf16* Wbase = Wfrag + ((size_t)(wave * TPW) * KBLK << 9) + (lane << 4);

#pragma unroll 1
      for (int kb = 0; kb < KBLK; ++kb) {
        union { v16bf v; v8bf h[2]; } au;
        au.h[0] = *(const v8bf*)(Arow + kb * 32 + khalf);
        au.h[1] = *(const v8bf*)(Arow + kb * 32 + khalf + 16);

        v16bf bfr[TPW];
#pragma unroll
        for (int q = 0; q < TPW; ++q) {
          const __bf16* bp = Wbase + ((size_t)(q * KBLK + kb) << 9);
          bfr[q] = *(const v16bf*)bp;
          // Unconditional speculative prefetch of this tile's next k-block
          // fragment (tail lands in the adjacent mapped ws region; harmless).
          __builtin_prefetch(bp + 512, 0, 1);
        }
#pragma unroll
        for (int q = 0; q < TPW; ++q)
          acc[q] = __builtin_amdgcn_wmma_f32_16x16x32_bf16(
              false, au.v, false, bfr[q], (short)0, acc[q], false, false);
      }
      // D layout: col n = tile*16 + (lane&15); row m = j + 8*(lane>>4)
#pragma unroll
      for (int q = 0; q < TPW; ++q) {
        const int tile = wave * TPW + q;
        const int n  = tile * 16 + (lane & 15);
        const int mb = (lane >> 4) << 3;
#pragma unroll
        for (int j = 0; j < 8; ++j) sG[(mb + j) * Gn + n] = acc[q][j];
      }
    }
    __syncthreads();

    // (c) LSTM pointwise: i,f,g,o split; update s, h; refresh h half of A
    for (int i = tid; i < B16 * Hn; i += NTHREADS) {
      int m = i >> 9, j = i & 511;
      float gi = sG[m * Gn + j]        + sBias[j];
      float gf = sG[m * Gn + 512 + j]  + sBias[512 + j];
      float gg = sG[m * Gn + 1024 + j] + sBias[1024 + j];
      float go = sG[m * Gn + 1536 + j] + sBias[1536 + j];
      float sv = sigm(gf) * sS[m][j] + sigm(gi) * tanhf(gg);
      float hv = sigm(go) * tanhf(sv);
      sS[m][j] = sv; sH[m][j] = hv;
      sA[m][512 + j] = (__bf16)hv;
    }
    __syncthreads();

    // (d) interface projections from h (gates buffer now reusable as scratch)
    for (int id = tid; id < 4096 + 3072 + 16 * 23; id += NTHREADS) {
      if (id < 4096) {                       // read keys rk[b][w][r]
        int m = id >> 8, rem = id & 255, w = rem >> 2, r = rem & 3;
        float d = dot512(Wrk + ((size_t)r * Wn + w) * Hn, &sH[m][0]);
        scr[RK_OFF + (m * 64 + w) * 4 + r] = d + brk[r * Wn + w];
      } else if (id < 7168) {                // wk / erase / wv
        int sub = id - 4096, which = sub >> 10, rem = sub & 1023;
        int m = rem >> 6, w = rem & 63;
        if (which == 0)
          scr[WK_OFF + m * 64 + w] = dot512(Wwk + (size_t)w * Hn, &sH[m][0]) + bwk[w];
        else if (which == 1)
          scr[ER_OFF + m * 64 + w] = sigm(dot512(We + (size_t)w * Hn, &sH[m][0]) + be[w]);
        else
          scr[WV_OFF + m * 64 + w] = sigm(dot512(Wv + (size_t)w * Hn, &sH[m][0]) + bv[w]);
      } else {                               // per-row scalars / small heads
        int sub = id - 7168, m = sub / 23, u = sub % 23;
        const float* h = &sH[m][0];
        if (u < 4)       sFG[m][u] = sigm(dot512(Wfg + (size_t)u * Hn, h) + bfg[u]);
        else if (u < 8)  { int r = u - 4; sRS[m][r] = oneplus(dot512(Wrs + (size_t)r * Hn, h) + brs[r]); }
        else if (u < 20) { int v = u - 8, mode = v >> 2, r = v & 3;
                           sRM[m][mode][r] = dot512(Wrm + ((size_t)r * 3 + mode) * Hn, h) + brm[r * 3 + mode]; }
        else if (u == 20) sWSs[m] = oneplus(dot512(Wws, h) + bws[0]);
        else if (u == 21) sAG[m]  = sigm(dot512(Wag, h) + bag[0]);
        else              sWGt[m] = sigm(dot512(Wwg, h) + bwg[0]);
      }
    }
    __syncthreads();

    // (d2) rm softmax, key norms, usage update (uses OLD ww + OLD rw)
    for (int id = tid; id < 64 + 80 + 2048; id += NTHREADS) {
      if (id < 64) {                         // softmax over 3 read modes
        int m = id >> 2, r = id & 3;
        float l0 = sRM[m][0][r], l1 = sRM[m][1][r], l2 = sRM[m][2][r];
        float mx = fmaxf(l0, fmaxf(l1, l2));
        float e0 = expf(l0 - mx), e1 = expf(l1 - mx), e2 = expf(l2 - mx);
        float inv = 1.f / (e0 + e1 + e2);
        sRM[m][0][r] = e0 * inv; sRM[m][1][r] = e1 * inv; sRM[m][2][r] = e2 * inv;
      } else if (id < 144) {                 // L2 norms of the 5 keys
        int q = id - 64, m = q / 5, k = q % 5;
        float s = 0.f;
        if (k < 4) for (int w = 0; w < Wn; ++w) { float v = scr[RK_OFF + (m * 64 + w) * 4 + k]; s += v * v; }
        else       for (int w = 0; w < Wn; ++w) { float v = scr[WK_OFF + m * 64 + w];           s += v * v; }
        sKN[m][k] = sqrtf(s);
      } else {                               // psi + usage
        int q = id - 144, m = q >> 7, n = q & 127;
        float psi = 1.f;
#pragma unroll
        for (int r = 0; r < Rn; ++r) psi *= (1.f - sFG[m][r] * sRW[m][n][r]);
        float uo = sU[m][n], wwo = sWW[m][n];
        float un = (uo + wwo - uo * wwo) * psi;
        sU[m][n] = un;
        scr[SU_OFF + m * 128 + n] = un;
      }
    }
    __syncthreads();

    // (e) bitonic sort of usage per row (ascending)
    for (int ksz = 2; ksz <= Nn; ksz <<= 1) {
      for (int jsz = ksz >> 1; jsz > 0; jsz >>= 1) {
        for (int e = tid; e < B16 * Nn; e += NTHREADS) {
          int m = e >> 7, i = e & 127, l = i ^ jsz;
          if (l > i) {
            float va = scr[SU_OFF + m * 128 + i], vb = scr[SU_OFF + m * 128 + l];
            bool up = ((i & ksz) == 0);
            if ((va > vb) == up) {
              scr[SU_OFF + m * 128 + i] = vb;
              scr[SU_OFF + m * 128 + l] = va;
            }
          }
        }
        __syncthreads();
      }
    }

    // (f) allocation (sorted-order, as in source) + write content scores (OLD mem)
    for (int id = tid; id < 16 + 2048; id += NTHREADS) {
      if (id < 16) {
        int m = id; float cum = 1.f;
        for (int i = 0; i < Nn; ++i) {
          float sv = scr[SU_OFF + m * 128 + i];
          scr[AL_OFF + m * 128 + i] = (1.f - sv) * cum;
          cum *= sv;
        }
      } else {
        int q = id - 16, m = q >> 7, n = q & 127;
        const float* mr = memB + ((size_t)(m * Nn + n)) * Wn;
        float d = 0.f, nm = 0.f;
        for (int w = 0; w < Wn; ++w) { float mv = mr[w]; d += mv * scr[WK_OFF + m * 64 + w]; nm += mv * mv; }
        scr[SC_OFF + (m * 128 + n) * 5 + 4] = sWSs[m] * d / ((sqrtf(nm) + 1e-6f) * sKN[m][4]);
      }
    }
    __syncthreads();

    // (g) write softmax over N -> cw (in place)
    for (int m = tid; m < B16; m += NTHREADS) {
      float mx = -3.4e38f;
      for (int n = 0; n < Nn; ++n) mx = fmaxf(mx, scr[SC_OFF + (m * 128 + n) * 5 + 4]);
      float sum = 0.f;
      for (int n = 0; n < Nn; ++n) { float e = expf(scr[SC_OFF + (m * 128 + n) * 5 + 4] - mx);
                                     scr[SC_OFF + (m * 128 + n) * 5 + 4] = e; sum += e; }
      float inv = 1.f / sum;
      for (int n = 0; n < Nn; ++n) scr[SC_OFF + (m * 128 + n) * 5 + 4] *= inv;
    }
    __syncthreads();

    // (h) new write weights
    for (int id = tid; id < B16 * Nn; id += NTHREADS) {
      int m = id >> 7, n = id & 127;
      sWW[m][n] = sWGt[m] * (sAG[m] * scr[AL_OFF + m * 128 + n] +
                             (1.f - sAG[m]) * scr[SC_OFF + (m * 128 + n) * 5 + 4]);
    }
    __syncthreads();

    // (i) memory write + ww row sums
    for (int id = tid; id < B16 * Nn * Wn + 16; id += NTHREADS) {
      if (id < B16 * Nn * Wn) {
        int m = id >> 13, rem = id & 8191, n = rem >> 6, w = rem & 63;
        memB[((size_t)(m * Nn + n)) * Wn + w] +=
            sWW[m][n] * (scr[WV_OFF + m * 64 + w] - scr[ER_OFF + m * 64 + w]);
      } else {
        int m = id - B16 * Nn * Wn;
        float s = 0.f;
        for (int n = 0; n < Nn; ++n) s += sWW[m][n];
        sWSum[m] = s;
      }
    }
    __syncthreads();

    // (j) precedence
    for (int id = tid; id < B16 * Nn; id += NTHREADS) {
      int m = id >> 7, n = id & 127;
      sP[m][n] = (1.f - sWSum[m]) * sP[m][n] + sWW[m][n];
    }
    __syncthreads();

    // (k) link matrix update + read content scores (NEW mem)
    for (int id = tid; id < B16 * Nn * Nn + B16 * Nn; id += NTHREADS) {
      if (id < B16 * Nn * Nn) {
        int m = id >> 14, rest = id & 16383, i = rest >> 7, j = rest & 127;
        size_t o = ((size_t)(m * Nn + i)) * Nn + j;
        float L = linkB[o];
        L = (1.f - sWW[m][i] - sWW[m][j]) * L + sWW[m][i] * sP[m][j];
        linkB[o] = (i == j) ? 0.f : L;
      } else {
        int q = id - B16 * Nn * Nn, m = q >> 7, n = q & 127;
        const float* mr = memB + ((size_t)(m * Nn + n)) * Wn;
        float nm = 0.f, d0 = 0.f, d1 = 0.f, d2 = 0.f, d3 = 0.f;
        for (int w = 0; w < Wn; ++w) {
          float mv = mr[w]; nm += mv * mv;
          const float* rkp = scr + RK_OFF + (m * 64 + w) * 4;
          d0 += mv * rkp[0]; d1 += mv * rkp[1]; d2 += mv * rkp[2]; d3 += mv * rkp[3];
        }
        float inv = 1.f / (sqrtf(nm) + 1e-6f);
        scr[SC_OFF + (m * 128 + n) * 5 + 0] = sRS[m][0] * d0 * inv / sKN[m][0];
        scr[SC_OFF + (m * 128 + n) * 5 + 1] = sRS[m][1] * d1 * inv / sKN[m][1];
        scr[SC_OFF + (m * 128 + n) * 5 + 2] = sRS[m][2] * d2 * inv / sKN[m][2];
        scr[SC_OFF + (m * 128 + n) * 5 + 3] = sRS[m][3] * d3 * inv / sKN[m][3];
      }
    }
    __syncthreads();

    // (l) read softmaxes over N (per head)
    for (int id = tid; id < B16 * Rn; id += NTHREADS) {
      int m = id >> 2, r = id & 3;
      float mx = -3.4e38f;
      for (int n = 0; n < Nn; ++n) mx = fmaxf(mx, scr[SC_OFF + (m * 128 + n) * 5 + r]);
      float sum = 0.f;
      for (int n = 0; n < Nn; ++n) { float e = expf(scr[SC_OFF + (m * 128 + n) * 5 + r] - mx);
                                     scr[SC_OFF + (m * 128 + n) * 5 + r] = e; sum += e; }
      float inv = 1.f / sum;
      for (int n = 0; n < Nn; ++n) scr[SC_OFF + (m * 128 + n) * 5 + r] *= inv;
    }
    __syncthreads();

    // (m) forward/backward weights + new read weights (uses OLD sRW)
    for (int id = tid; id < B16 * Nn; id += NTHREADS) {
      int m = id >> 7, n = id & 127;
      float fwd[4] = {0.f, 0.f, 0.f, 0.f}, bwd[4] = {0.f, 0.f, 0.f, 0.f};
      const float* Lrow = linkB + ((size_t)(m * Nn + n)) * Nn;
      const float* Lcol = linkB + ((size_t)(m * Nn)) * Nn + n;
      for (int j = 0; j < Nn; ++j) {
        float Lij = Lrow[j], Lji = Lcol[(size_t)j * Nn];
#pragma unroll
        for (int r = 0; r < Rn; ++r) {
          float rwj = sRW[m][j][r];
          fwd[r] += Lij * rwj; bwd[r] += Lji * rwj;
        }
      }
#pragma unroll
      for (int r = 0; r < Rn; ++r)
        scr[RWN_OFF + (m * 128 + n) * 4 + r] =
            sRM[m][0][r] * bwd[r] +
            sRM[m][1][r] * scr[SC_OFF + (m * 128 + n) * 5 + r] +
            sRM[m][2][r] * fwd[r];
    }
    __syncthreads();

    // (n) read vectors (NEW mem x new rw) + commit rw carry
    for (int id = tid; id < 4096 + B16 * Nn * Rn; id += NTHREADS) {
      if (id < 4096) {
        int m = id >> 8, rem = id & 255, w = rem >> 2, r = rem & 3;
        float acc = 0.f;
        for (int n = 0; n < Nn; ++n)
          acc += memB[((size_t)(m * Nn + n)) * Wn + w] * scr[RWN_OFF + (m * 128 + n) * 4 + r];
        scr[RV_OFF + m * 256 + w * 4 + r] = acc;   // r layout: w*R + r (matches reshape)
      } else {
        int q = id - 4096, m = q >> 9, rem = q & 511, n = rem >> 2, r = rem & 3;
        sRW[m][n][r] = scr[RWN_OFF + (m * 128 + n) * 4 + r];
      }
    }
    __syncthreads();

    // (o) emit hr = [h | r]; refresh r half of A for next step
    for (int id = tid; id < B16 * OUTC; id += NTHREADS) {
      int m = id / OUTC, c = id % OUTC;
      float v = (c < Hn) ? sH[m][c] : scr[RV_OFF + m * 256 + (c - Hn)];
      out[((size_t)t * Bn + (b0 + m)) * OUTC + c] = v;
      if (t == Tn - 1) out[((size_t)Tn * Bn + (b0 + m)) * OUTC + c] = v;
      if (c >= Hn) sA[m][c - Hn + Dn] = (__bf16)v;   // cols 256..511 = r
    }
    __syncthreads();
  }
}

extern "C" void kernel_launch(void* const* d_in, const int* in_sizes, int n_in,
                              void* d_out, int out_size, void* d_ws, size_t ws_size,
                              hipStream_t stream) {
  const float* x    = (const float*)d_in[0];
  const float* M0   = (const float*)d_in[1];
  const float* W_ih = (const float*)d_in[2];
  const float* b_ih = (const float*)d_in[3];
  const float* W_hh = (const float*)d_in[4];
  const float* b_hh = (const float*)d_in[5];
  const float* Wrk  = (const float*)d_in[6];
  const float* brk  = (const float*)d_in[7];
  const float* Wrs  = (const float*)d_in[8];
  const float* brs  = (const float*)d_in[9];
  const float* Wwk  = (const float*)d_in[10];
  const float* bwk  = (const float*)d_in[11];
  const float* Wws  = (const float*)d_in[12];
  const float* bws  = (const float*)d_in[13];
  const float* We   = (const float*)d_in[14];
  const float* be   = (const float*)d_in[15];
  const float* Wv   = (const float*)d_in[16];
  const float* bv   = (const float*)d_in[17];
  const float* Wfg  = (const float*)d_in[18];
  const float* bfg  = (const float*)d_in[19];
  const float* Wag  = (const float*)d_in[20];
  const float* bag  = (const float*)d_in[21];
  const float* Wwg  = (const float*)d_in[22];
  const float* bwg  = (const float*)d_in[23];
  const float* Wrm  = (const float*)d_in[24];
  const float* brm  = (const float*)d_in[25];

  char* ws = (char*)d_ws;
  __bf16* Wfrag = (__bf16*)(ws + WS_WFRAG_OFF);
  float*  memG  = (float*)(ws + WS_MEM_OFF);
  float*  linkG = (float*)(ws + WS_LINK_OFF);
  float*  out   = (float*)d_out;

  dnc_prep<<<(Gn * Kn + 255) / 256, 256, 0, stream>>>(W_ih, W_hh, Wfrag);
  dnc_forward<<<NWG, NTHREADS, 0, stream>>>(
      x, M0, b_ih, b_hh, Wrk, brk, Wrs, brs, Wwk, bwk, Wws, bws, We, be, Wv, bv,
      Wfg, bfg, Wag, bag, Wwg, bwg, Wrm, brm, Wfrag, memG, linkG, out);
}
